// PointTransformerLayer_29970281791913
// MI455X (gfx1250) — compile-verified, matrix operand order checked by reading, not run on hardware
//
#include <hip/hip_runtime.h>
#include <hip/hip_bf16.h>

// Shapes (fixed by the reference)
#define BB    4
#define NN    4096
#define CIN   64
#define COUTC 64
#define KNB   16
#define NH    4
#define HD    16
#define NROW  (BB * NN)   // 16384 total points

// Attention constant block (staged to LDS once per workgroup)
//   [    0, 8192)  wvpk  : Wv^T  packed f16 WMMA-B tiles
//   [ 8192,16384)  wp2pk : Wp2^T packed f16 WMMA-B tiles
//   [16384,17408)  WkM   : f32[64*4]  head-mean K weights
//   [17408,18176)  Wp1c  : f32[64*3]
//   [18176,18432)  bp1c  : f32[64]
//   [18432,18688)  bp2c  : f32[64]
//   [18688,18944)  bvc   : f32[64]
//   [18944,18960)  bkMc  : f32[4]
//   [18960,19200)  pad
#define ATTNBLK_BYTES 19200
#define ATTNBLK_DW    4800

typedef _Float16 v8h  __attribute__((ext_vector_type(8)));
typedef _Float16 v16h __attribute__((ext_vector_type(16)));
typedef float    v8f  __attribute__((ext_vector_type(8)));
typedef unsigned int u32x4 __attribute__((ext_vector_type(4)));
typedef int          i32x4 __attribute__((ext_vector_type(4)));
typedef int          i32x8 __attribute__((ext_vector_type(8)));

// K-slot mapping inside a 16x32 f16 WMMA A tile (ISA 7.12.2):
// lane group g = lane>>4; halves j=0..7 -> K = g*8 + j ; j=8..15 -> K = 16 + g*8 + (j-8)
__device__ __forceinline__ int kOf(int g, int j) {
    return (j < 8) ? (g * 8 + j) : (16 + g * 8 + (j - 8));
}

__device__ __forceinline__ float bperm_f(int srcLane, float v) {
    return __int_as_float(__builtin_amdgcn_ds_bpermute(srcLane << 2, __float_as_int(v)));
}

__device__ __forceinline__ v8f wmma16(v16h a, v16h b, v8f c) {
    // D = A(16x32 f16) * B(32x16 f16) + C(16x16 f32)
    return __builtin_amdgcn_wmma_f32_16x16x32_f16(false, a, false, b, (short)0, c,
                                                  false, false);
}

// ---------------------------------------------------------------------------
// prep: pack Wv^T / Wp2^T / Wfc^T into WMMA-B lane layout (f16), build
// head-mean matrices WkM/WqM (64x4), mean biases, and gather everything the
// attention kernel needs into one contiguous TDM-loadable block.
// ---------------------------------------------------------------------------
__global__ void pt_prep_kernel(const float* __restrict__ Wq, const float* __restrict__ bq,
                               const float* __restrict__ Wk, const float* __restrict__ bk,
                               const float* __restrict__ Wv, const float* __restrict__ bv,
                               const float* __restrict__ Wp1, const float* __restrict__ bp1,
                               const float* __restrict__ Wp2, const float* __restrict__ bp2,
                               const float* __restrict__ Wfc,
                               char* __restrict__ attnblk, _Float16* __restrict__ wfcpk,
                               float* __restrict__ WqM, float* __restrict__ bqM) {
    _Float16* wvpk  = (_Float16*)(attnblk);
    _Float16* wp2pk = (_Float16*)(attnblk + 8192);
    float*    WkM   = (float*)(attnblk + 16384);
    float*    wp1c  = (float*)(attnblk + 17408);
    float*    bp1c  = (float*)(attnblk + 18176);
    float*    bp2c  = (float*)(attnblk + 18432);
    float*    bvc   = (float*)(attnblk + 18688);
    float*    bkMc  = (float*)(attnblk + 18944);
    float*    padc  = (float*)(attnblk + 18960);

    int tid  = threadIdx.x;            // 256 threads
    int tile = tid >> 5;               // 0..7 : tile = cb*4 + nb
    int lane = tid & 31;
    int cb   = tile >> 2;              // K-block (0,1)
    int nb   = tile & 3;               // output col block == head
    int g    = lane >> 4;
    int col  = nb * 16 + (lane & 15);  // output column (== W row)
    for (int j = 0; j < 16; ++j) {
        int kg = cb * 32 + kOf(g, j);  // input channel
        int o  = (tile * 32 + lane) * 16 + j;
        wvpk[o]  = (_Float16)Wv[col * CIN + kg];
        wp2pk[o] = (_Float16)Wp2[col * COUTC + kg];
        wfcpk[o] = (_Float16)Wfc[col * COUTC + kg];
    }
    // head-mean weight matrices: WxM[c][h] = (1/16) sum_d Wx[h*16+d][c]
    int c = tid >> 2, h = tid & 3;
    float sk = 0.f, sq = 0.f;
    for (int d = 0; d < HD; ++d) {
        sk += Wk[(h * HD + d) * CIN + c];
        sq += Wq[(h * HD + d) * CIN + c];
    }
    WkM[c * 4 + h] = sk * (1.f / 16.f);
    WqM[c * 4 + h] = sq * (1.f / 16.f);
    if (tid < 4) {
        float s = 0.f;
        for (int d = 0; d < HD; ++d) s += bk[tid * HD + d];
        bkMc[tid] = s * (1.f / 16.f);
    } else if (tid < 8) {
        int h2 = tid - 4;
        float s = 0.f;
        for (int d = 0; d < HD; ++d) s += bq[h2 * HD + d];
        bqM[h2] = s * (1.f / 16.f);
    }
    if (tid < 192) wp1c[tid] = Wp1[tid];
    if (tid < 64) {
        bp1c[tid] = bp1[tid];
        bp2c[tid] = bp2[tid];
        bvc[tid]  = bv[tid];
    }
    if (tid < 60) padc[tid] = 0.f;
}

// ---------------------------------------------------------------------------
// x -> f16 (feeds WMMA A tiles via gather)
// ---------------------------------------------------------------------------
__global__ void pt_x2h_kernel(const float* __restrict__ x, _Float16* __restrict__ x16) {
    int i  = blockIdx.x * 256 + threadIdx.x;   // grid 4096 covers NROW*64
    x16[i] = (_Float16)x[i];
}

// ---------------------------------------------------------------------------
// qmean[r][h] = mean_d (x[r] @ Wq^T + bq)[h*16+d]  (linear -> x @ WqM + bqM)
// ---------------------------------------------------------------------------
__global__ void pt_qmean_kernel(const float* __restrict__ x, const float* __restrict__ WqM,
                                const float* __restrict__ bqM, float* __restrict__ qm) {
    int r = blockIdx.x * 256 + threadIdx.x;    // 64 blocks -> 16384 rows
    float a0 = 0.f, a1 = 0.f, a2 = 0.f, a3 = 0.f;
    for (int c = 0; c < CIN; ++c) {
        float xv = x[r * CIN + c];
        a0 += xv * WqM[c * 4 + 0];
        a1 += xv * WqM[c * 4 + 1];
        a2 += xv * WqM[c * 4 + 2];
        a3 += xv * WqM[c * 4 + 3];
    }
    qm[r * 4 + 0] = a0 + bqM[0];
    qm[r * 4 + 1] = a1 + bqM[1];
    qm[r * 4 + 2] = a2 + bqM[2];
    qm[r * 4 + 3] = a3 + bqM[3];
}

// ---------------------------------------------------------------------------
// kNN: one thread per query, LDS-tiled candidates, register top-16 (ascending
// squared distance == descending neg_sqdist of the reference top_k).
// ---------------------------------------------------------------------------
__global__ void pt_knn_kernel(const float* __restrict__ pos, int* __restrict__ idx) {
    __shared__ float sx[256], sy[256], sz[256];
    int t    = threadIdx.x;
    int r    = blockIdx.x * 256 + t;
    int base = (r >> 12) << 12;     // batch base (N = 4096)
    float qx = pos[r * 3 + 0], qy = pos[r * 3 + 1], qz = pos[r * 3 + 2];
    float bd[16];
    int   bi[16];
#pragma unroll
    for (int i = 0; i < 16; ++i) { bd[i] = 3.4e38f; bi[i] = 0; }
    for (int m0 = 0; m0 < NN; m0 += 256) {
        __syncthreads();
        int mr = base + m0 + t;
        sx[t] = pos[mr * 3 + 0];
        sy[t] = pos[mr * 3 + 1];
        sz[t] = pos[mr * 3 + 2];
        __syncthreads();
        for (int j = 0; j < 256; ++j) {
            float dx = qx - sx[j], dy = qy - sy[j], dz = qz - sz[j];
            float d  = dx * dx + dy * dy + dz * dz;
            if (d < bd[15]) {
                bd[15] = d;
                bi[15] = m0 + j;
#pragma unroll
                for (int s = 15; s > 0; --s) {
                    if (bd[s] < bd[s - 1]) {
                        float td = bd[s]; bd[s] = bd[s - 1]; bd[s - 1] = td;
                        int   ti = bi[s]; bi[s] = bi[s - 1]; bi[s - 1] = ti;
                    }
                }
            }
        }
    }
#pragma unroll
    for (int i = 0; i < 16; ++i) idx[r * 16 + i] = bi[i];
}

// ---------------------------------------------------------------------------
// Fused attention: one wave32 per query; K=16 neighbors = WMMA A rows.
// Weights/constants staged to LDS once per workgroup via the Tensor Data
// Mover (tensor_load_to_lds + s_wait_tensorcnt), WMMA B tiles fed from LDS.
// ---------------------------------------------------------------------------
__global__ void pt_attn_kernel(const _Float16* __restrict__ x16, const float* __restrict__ pos,
                               const int* __restrict__ idx, const float* __restrict__ qm,
                               const char* __restrict__ attnblk,
                               _Float16* __restrict__ agg16) {
    __shared__ __align__(16) unsigned char smem[ATTNBLK_BYTES];

    // ---- stage the 19200-B weight/constant block into LDS
#if __has_builtin(__builtin_amdgcn_tensor_load_to_lds)
    if (threadIdx.x < 32) {
        unsigned long long ga = (unsigned long long)attnblk;
        unsigned int lds_off  = (unsigned int)(unsigned long long)(uintptr_t)&smem[0];
        u32x4 g0;
        g0[0] = 1u;                                              // count=1, user mode
        g0[1] = lds_off;                                         // lds_addr
        g0[2] = (unsigned int)ga;                                // global_addr[31:0]
        g0[3] = (unsigned int)((ga >> 32) & 0x01ffffffu) | (2u << 30);  // addr hi | type=2
        i32x8 g1;
        g1[0] = (2 << 16);                                       // data_size = 4B
        g1[1] = (int)((ATTNBLK_DW & 0xffff) << 16);              // tensor_dim0 lo16
        g1[2] = (int)((ATTNBLK_DW >> 16) | (1 << 16));           // dim0 hi | tensor_dim1=1
        g1[3] = (int)(ATTNBLK_DW << 16);                         // tile_dim0
        g1[4] = 0;                                               // tile_dim1/2 unused
        g1[5] = ATTNBLK_DW;                                      // tensor_dim0_stride
        g1[6] = 0;
        g1[7] = 0;
        i32x4 gz = {0, 0, 0, 0};
#if __clang_major__ >= 23
        i32x8 gz8 = {0, 0, 0, 0, 0, 0, 0, 0};
        __builtin_amdgcn_tensor_load_to_lds(g0, g1, gz, gz, gz8, 0);
#else
        __builtin_amdgcn_tensor_load_to_lds(g0, g1, gz, gz, 0);
#endif
        __builtin_amdgcn_s_wait_tensorcnt(0);
    }
#else
    for (int i = threadIdx.x; i < ATTNBLK_DW; i += 256)
        ((float*)smem)[i] = ((const float*)attnblk)[i];
#endif
    __syncthreads();

    const _Float16* s_wv  = (const _Float16*)(smem);
    const _Float16* s_wp2 = (const _Float16*)(smem + 8192);
    const float*    s_wkm = (const float*)(smem + 16384);
    const float*    s_wp1 = (const float*)(smem + 17408);
    const float*    s_bp1 = (const float*)(smem + 18176);
    const float*    s_bp2 = (const float*)(smem + 18432);
    const float*    s_bv  = (const float*)(smem + 18688);
    const float*    s_bkm = (const float*)(smem + 18944);

    int lane = threadIdx.x & 31;
    int wid  = threadIdx.x >> 5;
    int q    = blockIdx.x * 8 + wid;     // global query row
    int g    = lane >> 4;
    int m    = lane & 15;                // neighbor row handled by this lane (A side)
    int base = (q >> 12) << 12;          // batch base

    int nbr = idx[q * 16 + m];
    int xr  = base + nbr;
    __builtin_prefetch(&x16[xr * 64], 0, 3);   // global_prefetch_b8

    // ---- A tiles of gathered neighbor features (two 16-byte loads per half-tile)
    const v8h* xp = (const v8h*)(x16 + xr * 64);
    v8h  l0 = xp[g],     h0 = xp[2 + g];
    v8h  l1 = xp[4 + g], h1 = xp[6 + g];
    v16h ax0 = __builtin_shufflevector(l0, h0, 0,1,2,3,4,5,6,7,8,9,10,11,12,13,14,15);
    v16h ax1 = __builtin_shufflevector(l1, h1, 0,1,2,3,4,5,6,7,8,9,10,11,12,13,14,15);

    // ---- kmean[m][h] (x_nb @ WkM): each lane covers half the channels, xor-16 merge
    float km[4] = {0.f, 0.f, 0.f, 0.f};
#pragma unroll
    for (int j = 0; j < 16; ++j) {
        int   k0 = kOf(g, j);
        float a0 = (float)ax0[j];
        float a1 = (float)ax1[j];
#pragma unroll
        for (int h = 0; h < 4; ++h)
            km[h] += a0 * s_wkm[k0 * 4 + h] + a1 * s_wkm[(32 + k0) * 4 + h];
    }
#pragma unroll
    for (int h = 0; h < 4; ++h) {
        km[h] += bperm_f(lane ^ 16, km[h]);
        km[h] += s_bkm[h];
    }

    // ---- pos-enc layer 1 (3->64, relu) computed directly into the A-tile slots
    float dx = pos[q * 3 + 0] - pos[xr * 3 + 0];
    float dy = pos[q * 3 + 1] - pos[xr * 3 + 1];
    float dz = pos[q * 3 + 2] - pos[xr * 3 + 2];
    v16h ap0, ap1;
#pragma unroll
    for (int j = 0; j < 16; ++j) {
        int   c0 = kOf(g, j);
        int   c1 = 32 + c0;
        float t0 = dx * s_wp1[c0 * 3 + 0] + dy * s_wp1[c0 * 3 + 1] + dz * s_wp1[c0 * 3 + 2] + s_bp1[c0];
        float t1 = dx * s_wp1[c1 * 3 + 0] + dy * s_wp1[c1 * 3 + 1] + dz * s_wp1[c1 * 3 + 2] + s_bp1[c1];
        ap0[j] = (_Float16)fmaxf(t0, 0.f);
        ap1[j] = (_Float16)fmaxf(t1, 0.f);
    }

    float qmv[4];
#pragma unroll
    for (int h = 0; h < 4; ++h) qmv[h] = qm[q * 4 + h];

    // ---- per-head tiles: head h == output col-block nb (D == 16 == tile width)
#pragma unroll
    for (int nb = 0; nb < 4; ++nb) {
        v16h b0 = *(const v16h*)(s_wv  + ((0 * 4 + nb) * 32 + lane) * 16);
        v16h b1 = *(const v16h*)(s_wv  + ((1 * 4 + nb) * 32 + lane) * 16);
        v16h p0 = *(const v16h*)(s_wp2 + ((0 * 4 + nb) * 32 + lane) * 16);
        v16h p1 = *(const v16h*)(s_wp2 + ((1 * 4 + nb) * 32 + lane) * 16);
        v8f vac = {};
        v8f pac = {};
        vac = wmma16(ax0, b0, vac);
        vac = wmma16(ax1, b1, vac);
        pac = wmma16(ap0, p0, pac);
        pac = wmma16(ap1, p1, pac);

        int   col = nb * 16 + m;
        float bva = s_bv[col], bpa = s_bp2[col];
        float logit[8];
#pragma unroll
        for (int v = 0; v < 8; ++v) {
            vac[v] += bva;
            pac[v] += bpa;
            // pemean over the 16 lanes of this half (C/D layout: col = lane&15)
            float s = pac[v];
            s += bperm_f(lane ^ 1, s);
            s += bperm_f(lane ^ 2, s);
            s += bperm_f(lane ^ 4, s);
            s += bperm_f(lane ^ 8, s);
            float pmean = s * (1.f / 16.f);
            // kmean lives in lane (row); rows in this half are v + 8*g
            logit[v] = qmv[nb] - bperm_f(v + 8 * g, km[nb]) + pmean;
        }
        // softmax over the 16 neighbors (8 regs x 2 lane-groups)
        float mx = logit[0];
#pragma unroll
        for (int v = 1; v < 8; ++v) mx = fmaxf(mx, logit[v]);
        mx = fmaxf(mx, bperm_f(lane ^ 16, mx));
        float ev[8], ssum = 0.f;
#pragma unroll
        for (int v = 0; v < 8; ++v) {
            ev[v] = __expf(logit[v] - mx);
            ssum += ev[v];
        }
        ssum += bperm_f(lane ^ 16, ssum);
        float inv = 1.f / ssum;
        float agg = 0.f;
#pragma unroll
        for (int v = 0; v < 8; ++v) agg += ev[v] * (vac[v] + pac[v]);
        agg *= inv;
        agg += bperm_f(lane ^ 16, agg);
        if (lane < 16) agg16[q * 64 + col] = (_Float16)agg;
    }
}

// ---------------------------------------------------------------------------
// Final FC: out = agg @ Wfc^T + bfc + x   (WMMA GEMM, 16-row tiles per wave)
// ---------------------------------------------------------------------------
__global__ void pt_fc_kernel(const _Float16* __restrict__ agg16,
                             const _Float16* __restrict__ wfcpk,
                             const float* __restrict__ bfc, const float* __restrict__ x,
                             float* __restrict__ out) {
    int lane = threadIdx.x & 31;
    int wid  = threadIdx.x >> 5;
    int r0   = (blockIdx.x * 8 + wid) * 16;
    int g    = lane >> 4, m = lane & 15;
    const v8h* ap = (const v8h*)(agg16 + (r0 + m) * 64);
    v16h a0 = __builtin_shufflevector(ap[g],     ap[2 + g], 0,1,2,3,4,5,6,7,8,9,10,11,12,13,14,15);
    v16h a1 = __builtin_shufflevector(ap[4 + g], ap[6 + g], 0,1,2,3,4,5,6,7,8,9,10,11,12,13,14,15);
#pragma unroll
    for (int nb = 0; nb < 4; ++nb) {
        v16h b0 = *(const v16h*)(wfcpk + ((0 * 4 + nb) * 32 + lane) * 16);
        v16h b1 = *(const v16h*)(wfcpk + ((1 * 4 + nb) * 32 + lane) * 16);
        v8f acc = {};
        acc = wmma16(a0, b0, acc);
        acc = wmma16(a1, b1, acc);
        int   col  = nb * 16 + m;
        float bias = bfc[col];
#pragma unroll
        for (int v = 0; v < 8; ++v) {
            int row = r0 + v + 8 * g;
            out[row * 64 + col] = acc[v] + bias + x[row * 64 + col];
        }
    }
}

// ---------------------------------------------------------------------------
extern "C" void kernel_launch(void* const* d_in, const int* in_sizes, int n_in,
                              void* d_out, int out_size, void* d_ws, size_t ws_size,
                              hipStream_t stream) {
    const float* x   = (const float*)d_in[0];
    const float* pos = (const float*)d_in[1];
    const float* Wq  = (const float*)d_in[2];
    const float* bq  = (const float*)d_in[3];
    const float* Wk  = (const float*)d_in[4];
    const float* bk  = (const float*)d_in[5];
    const float* Wv  = (const float*)d_in[6];
    const float* bv  = (const float*)d_in[7];
    const float* Wp1 = (const float*)d_in[8];
    const float* bp1 = (const float*)d_in[9];
    const float* Wp2 = (const float*)d_in[10];
    const float* bp2 = (const float*)d_in[11];
    const float* Wfc = (const float*)d_in[12];
    const float* bfc = (const float*)d_in[13];
    float* out = (float*)d_out;

    char* ws = (char*)d_ws;
    int*      idx     = (int*)(ws + 0);             // 16384*16 int      (1 MiB)
    float*    qm      = (float*)(ws + 1048576);     // 16384*4  f32     (256 KiB)
    _Float16* x16     = (_Float16*)(ws + 1310720);  // 16384*64 f16       (2 MiB)
    _Float16* agg16   = (_Float16*)(ws + 3407872);  // 16384*64 f16       (2 MiB)
    char*     attnblk = ws + 5505024;               // 19200 B TDM block
    _Float16* wfcpk   = (_Float16*)(ws + 5524224);  // 8 KiB packed Wfc
    float*    WqM     = (float*)(ws + 5532416);     // 64*4 f32
    float*    bqM     = (float*)(ws + 5533440);     // 4 f32

    pt_prep_kernel<<<1, 256, 0, stream>>>(Wq, bq, Wk, bk, Wv, bv, Wp1, bp1, Wp2, bp2,
                                          Wfc, attnblk, wfcpk, WqM, bqM);
    pt_x2h_kernel<<<4096, 256, 0, stream>>>(x, x16);
    pt_qmean_kernel<<<64, 256, 0, stream>>>(x, WqM, bqM, qm);
    pt_knn_kernel<<<64, 256, 0, stream>>>(pos, idx);
    pt_attn_kernel<<<2048, 256, 0, stream>>>(x16, pos, idx, qm, attnblk, agg16);
    pt_fc_kernel<<<128, 256, 0, stream>>>(agg16, wfcpk, bfc, x, out);
}
